// LSTM_42322607734854
// MI455X (gfx1250) — matrix operand order repaired and checked by reading, not compiled, not run on hardware
//
#include <hip/hip_runtime.h>
#include <hip/hip_bf16.h>

// Problem constants (match reference)
#define BB 128
#define TT 256
#define DD 256
#define HH 1024
#define CC 10
#define KTOT 1280        // D + H (concatenated K for fused step GEMM)
#define NG   4096        // 4*H gate columns
#define KTILES 40        // KTOT / 32

typedef __attribute__((ext_vector_type(16))) __bf16 bf16x16;
typedef __attribute__((ext_vector_type(8)))  float  f32x8;

struct Bytes32 { uint4 lo, hi; };

// Load a 32-byte WMMA fragment from two (possibly disjoint) 16B segments.
__device__ __forceinline__ bf16x16 load_frag(const __bf16* p0, const __bf16* p1) {
    Bytes32 b;
    b.lo = *(const uint4*)p0;
    b.hi = *(const uint4*)p1;
    return __builtin_bit_cast(bf16x16, b);
}

__device__ __forceinline__ float sigm_f(float x) {
    return 1.0f / (1.0f + __expf(-x));
}
__device__ __forceinline__ float tanh_f(float x) {
    x = fminf(fmaxf(x, -15.0f), 15.0f);
    float e = __expf(2.0f * x);
    return (e - 1.0f) / (e + 1.0f);
}

// ---------------------------------------------------------------------------
// 1) Pack the 8 fp32 gate weight matrices into one bf16 buffer, pre-swizzled
//    into per-lane WMMA B-fragment order. Logical layout: Wcomb[K=1280][N=4096]
//    with rows 0..255 from W_{g,i,f,o}x and rows 256..1279 from W_{g,i,f,o}h,
//    columns [gate*1024 + j]. Fragment tile (kt, nt) stores 32 lanes x 16 bf16
//    contiguously (1 KB) so it can be moved/loaded as contiguous b128s.
// ---------------------------------------------------------------------------
__global__ void pack_w_kernel(const float* __restrict__ Wgx, const float* __restrict__ Wix,
                              const float* __restrict__ Wfx, const float* __restrict__ Wox,
                              const float* __restrict__ Wgh, const float* __restrict__ Wih,
                              const float* __restrict__ Wfh, const float* __restrict__ Woh,
                              __bf16* __restrict__ Wp) {
    long tid = (long)blockIdx.x * blockDim.x + threadIdx.x;
    if (tid >= (long)KTOT * NG) return;
    int k = (int)(tid / NG);
    int n = (int)(tid % NG);
    int gate = n >> 10;          // 0..3 : g,i,f,o (concat along columns)
    int cg   = n & 1023;         // column within gate

    float v;
    if (k < DD) {
        const float* w = (gate == 0) ? Wgx : (gate == 1) ? Wix : (gate == 2) ? Wfx : Wox;
        v = w[(size_t)k * HH + cg];
    } else {
        const float* w = (gate == 0) ? Wgh : (gate == 1) ? Wih : (gate == 2) ? Wfh : Woh;
        v = w[(size_t)(k - DD) * HH + cg];
    }

    int kt  = k >> 5, kin = k & 31;
    int nt  = n >> 4, nin = n & 15;
    int lane = nin + ((kin & 16) ? 16 : 0);  // lanes 0-15: K 0..15, 16-31: K 16..31
    int kl   = kin & 15;                     // packed pairs: dword v holds K=2v,2v+1
    size_t dst = ((size_t)(nt * KTILES + kt) * 512) + (size_t)lane * 16 + (kl >> 1) * 2 + (kl & 1);
    Wp[dst] = (__bf16)v;
}

// ---------------------------------------------------------------------------
// 2) Embedding gather: xe[t][b][d] = bf16(emb[x[b,t]][d])   ([T,B,D] bf16)
// ---------------------------------------------------------------------------
__global__ void embed_kernel(const int* __restrict__ x, const float* __restrict__ emb,
                             __bf16* __restrict__ xe) {
    long tid = (long)blockIdx.x * blockDim.x + threadIdx.x;
    if (tid >= (long)TT * BB * DD) return;
    int t   = (int)(tid / (BB * DD));
    int rem = (int)(tid % (BB * DD));
    int b   = rem / DD;
    int d   = rem % DD;
    int xv  = x[(size_t)b * TT + t];                 // x is [B, T]
    xe[tid] = (__bf16)emb[(size_t)xv * DD + d];
}

// ---------------------------------------------------------------------------
// 3) Zero-init recurrent state
// ---------------------------------------------------------------------------
__global__ void zero_state_kernel(__bf16* __restrict__ hb0, float* __restrict__ hf,
                                  float* __restrict__ cf) {
    int i = blockIdx.x * blockDim.x + threadIdx.x;
    if (i < BB * HH) { hb0[i] = (__bf16)0.0f; hf[i] = 0.0f; cf[i] = 0.0f; }
}

// ---------------------------------------------------------------------------
// 4) One LSTM timestep, fully fused:
//    z = [x_emb_t | h_{t-1}] @ Wcomb + bias ; gates ; c,h update.
//    grid = 64 blocks (16 hidden cols each), block = 128 threads = 4 waves.
//    Each wave owns TWO 16-row M-tiles (32 batch rows) x 4 gate tiles ->
//    8 WMMAs per K-tile against one set of shared B fragments.
//    B fragments are staged cooperatively in LDS via gfx1250 async
//    global->LDS copies (ASYNCcnt), double-buffered so the copy for kt+1 is
//    in flight while kt computes. A fragments are pipelined one K-tile ahead.
// ---------------------------------------------------------------------------
__global__ __launch_bounds__(128) void lstm_step_kernel(
        const __bf16* __restrict__ xet,     // [B, D]  bf16, this timestep
        const __bf16* __restrict__ hb_in,   // [B, H]  bf16, h_{t-1}
        __bf16*       __restrict__ hb_out,  // [B, H]  bf16, h_t
        float*        __restrict__ hf,      // [B, H]  f32,  h_t
        float*        __restrict__ cf,      // [B, H]  f32,  c (in/out)
        const __bf16* __restrict__ Wp,      // packed bf16 weights
        const float*  __restrict__ bg, const float* __restrict__ bi,
        const float*  __restrict__ bfr, const float* __restrict__ bo) {
    // LDS stage: 2 buffers x 4 gate tiles x 512 bf16 (1 KB/tile) = 8 KB
    __shared__ __bf16 stage[2 * 4 * 512];

    const int lane  = threadIdx.x & 31;
    const int w     = threadIdx.x >> 5;   // 0..3
    const int m0    = w << 5;             // 32 batch rows per wave (2 M-tiles)
    const int j0    = blockIdx.x << 4;    // 16 hidden columns per block
    const int lrow  = lane & 15;
    const int lhalf = lane >> 4;
    const int ntb   = j0 >> 4;            // base n-tile (gate 0)

    // Cooperative B staging: wave w copies gate tile w (1 KB) as two
    // wave-wide async b128s (512 B each). ASYNCcnt is in-order, so a single
    // s_wait_asynccnt 0 covers both.
    auto issue_b = [&](int kt, int buf) {
        size_t tile_base = (size_t)(((w << 6) + ntb) * KTILES + kt) * 512;
        const __bf16* gsrc = Wp + tile_base + (size_t)lane * 8;
        const __bf16* ldst = &stage[buf * 2048 + w * 512 + lane * 8];
        unsigned l0 = (unsigned)(uintptr_t)ldst;    // addr[31:0] = LDS offset
        asm volatile("global_load_async_to_lds_b128 %0, %1, off"
                     :: "v"(l0), "v"(gsrc) : "memory");
        const __bf16* gsrc2 = gsrc + 256;
        unsigned l1 = l0 + 512;
        asm volatile("global_load_async_to_lds_b128 %0, %1, off"
                     :: "v"(l1), "v"(gsrc2) : "memory");
    };

    auto load_a = [&](int kt, int mrow) -> bf16x16 {
        const __bf16* src; int kstr, kk;
        if (kt < 8) { src = xet;   kstr = DD; kk = kt << 5; }
        else        { src = hb_in; kstr = HH; kk = (kt - 8) << 5; }
        // A k order used here (same as packed B): lanes 0-15 hold K 0..15 of their row,
        // lanes 16-31 hold K 16..31.
        const __bf16* ap = src + (size_t)(mrow + lrow) * kstr + kk + (lhalf << 4);
        return load_frag(ap, ap + 8);
    };

    f32x8 acc[8] = {};                    // [mtile 0|1][gate g,i,f,o]

    bf16x16 aN0 = load_a(0, m0);          // pipelined A fragments
    bf16x16 aN1 = load_a(0, m0 + 16);
    issue_b(0, 0);                        // pipelined B stage
    int buf = 0;

#pragma unroll 1
    for (int kt = 0; kt < KTILES; ++kt) {
        // My async copies for `buf` complete:
        asm volatile("s_wait_asynccnt 0x0" ::: "memory");
        // Everyone's copies complete AND everyone done reading the other buffer:
        __syncthreads();
        // Launch next tile's copies into the other buffer (fly during compute).
        if (kt + 1 < KTILES) issue_b(kt + 1, buf ^ 1);

        bf16x16 a0 = aN0, a1 = aN1;
        if (kt + 1 < KTILES) {            // overlaps with WMMAs below
            aN0 = load_a(kt + 1, m0);
            aN1 = load_a(kt + 1, m0 + 16);
        }

        // Load ALL four B fragments first (clauses the 8 ds_load_b128s),
        // then issue 8 back-to-back WMMAs.
        const __bf16* sb = &stage[buf * 2048];
        bf16x16 bm[4];
#pragma unroll
        for (int g = 0; g < 4; ++g) {
            const __bf16* bp = sb + g * 512 + lane * 16;
            bm[g] = load_frag(bp, bp + 8);
        }
#pragma unroll
        for (int g = 0; g < 4; ++g) {
            acc[g]     = __builtin_amdgcn_wmma_f32_16x16x32_bf16(
                false, a0, false, bm[g], (short)0, acc[g],     false, false);
            acc[4 + g] = __builtin_amdgcn_wmma_f32_16x16x32_bf16(
                false, a1, false, bm[g], (short)0, acc[4 + g], false, false);
        }
        buf ^= 1;
    }

    // Epilogue: bias + nonlinearities + state update, directly on C layout:
    // element r of lane L is (row mbase + r + 8*(L>=16), col j0 + (L&15)).
    const int col = j0 + lrow;
    const float biasG = bg[col], biasI = bi[col], biasF = bfr[col], biasO = bo[col];

#pragma unroll
    for (int s = 0; s < 2; ++s) {
        const int mbase = m0 + (s << 4);
#pragma unroll
        for (int r = 0; r < 8; ++r) {
            int row = mbase + r + (lhalf << 3);
            size_t idx = (size_t)row * HH + col;
            float gv = tanh_f(acc[s * 4 + 0][r] + biasG);
            float iv = sigm_f(acc[s * 4 + 1][r] + biasI);
            float fv = sigm_f(acc[s * 4 + 2][r] + biasF);
            float ov = sigm_f(acc[s * 4 + 3][r] + biasO);
            float cn = gv * iv + cf[idx] * fv;
            cf[idx] = cn;
            float hn = tanh_f(cn) * ov;
            hf[idx] = hn;
            hb_out[idx] = (__bf16)hn;
        }
    }
}

// ---------------------------------------------------------------------------
// 5) Head: p = h_T @ W_ph + b_p ; log_softmax.  [128x1024]@[1024x10] is tiny.
// ---------------------------------------------------------------------------
__global__ void head_kernel(const float* __restrict__ hf, const float* __restrict__ Wph,
                            const float* __restrict__ bp, float* __restrict__ out) {
    int b = threadIdx.x;            // 128 threads, one batch row each
    if (b >= BB) return;
    float p[CC];
#pragma unroll
    for (int c = 0; c < CC; ++c) p[c] = bp[c];
    const float* hr = hf + (size_t)b * HH;
    for (int j = 0; j < HH; ++j) {
        float hv = hr[j];
        const float* wr = Wph + (size_t)j * CC;
#pragma unroll
        for (int c = 0; c < CC; ++c) p[c] = fmaf(hv, wr[c], p[c]);
    }
    float m = p[0];
#pragma unroll
    for (int c = 1; c < CC; ++c) m = fmaxf(m, p[c]);
    float s = 0.0f;
#pragma unroll
    for (int c = 0; c < CC; ++c) s += __expf(p[c] - m);
    float lse = m + __logf(s);
#pragma unroll
    for (int c = 0; c < CC; ++c) out[(size_t)b * CC + c] = p[c] - lse;
}

// ---------------------------------------------------------------------------
extern "C" void kernel_launch(void* const* d_in, const int* in_sizes, int n_in,
                              void* d_out, int out_size, void* d_ws, size_t ws_size,
                              hipStream_t stream) {
    const int*   x    = (const int*)  d_in[0];
    const float* emb  = (const float*)d_in[1];
    const float* Wgx  = (const float*)d_in[2];
    const float* Wix  = (const float*)d_in[3];
    const float* Wfx  = (const float*)d_in[4];
    const float* Wox  = (const float*)d_in[5];
    const float* Wgh  = (const float*)d_in[6];
    const float* Wih  = (const float*)d_in[7];
    const float* Wfh  = (const float*)d_in[8];
    const float* Woh  = (const float*)d_in[9];
    const float* bg   = (const float*)d_in[10];
    const float* bi   = (const float*)d_in[11];
    const float* bfr  = (const float*)d_in[12];
    const float* bo   = (const float*)d_in[13];
    const float* Wph  = (const float*)d_in[14];
    const float* bp   = (const float*)d_in[15];
    float* out = (float*)d_out;

    // Workspace carve-up (all 256B-aligned slices).
    char* ws = (char*)d_ws;
    __bf16* Wp  = (__bf16*)ws;  ws += (size_t)KTOT * NG * 2;        // 10 MB packed weights
    __bf16* xe  = (__bf16*)ws;  ws += (size_t)TT * BB * DD * 2;     // 16 MB embedded x
    __bf16* hb0 = (__bf16*)ws;  ws += (size_t)BB * HH * 2;          // h bf16 ping
    __bf16* hb1 = (__bf16*)ws;  ws += (size_t)BB * HH * 2;          // h bf16 pong
    float*  hf  = (float*)ws;   ws += (size_t)BB * HH * 4;          // h f32 (for head)
    float*  cf  = (float*)ws;   ws += (size_t)BB * HH * 4;          // c f32

    // 1) pack weights fp32 -> bf16 WMMA-fragment order
    {
        long n = (long)KTOT * NG;
        pack_w_kernel<<<dim3((unsigned)((n + 255) / 256)), dim3(256), 0, stream>>>(
            Wgx, Wix, Wfx, Wox, Wgh, Wih, Wfh, Woh, Wp);
    }
    // 2) embedding gather -> bf16
    {
        long n = (long)TT * BB * DD;
        embed_kernel<<<dim3((unsigned)((n + 255) / 256)), dim3(256), 0, stream>>>(x, emb, xe);
    }
    // 3) zero state
    zero_state_kernel<<<dim3((BB * HH + 255) / 256), dim3(256), 0, stream>>>(hb0, hf, cf);

    // 4) 256 dependent timesteps (stream order gives the RAW chain; h is
    //    double-buffered so epilogue writes never race other blocks' reads).
    for (int t = 0; t < TT; ++t) {
        const __bf16* xet = xe + (size_t)t * BB * DD;
        const __bf16* hin  = (t & 1) ? hb1 : hb0;
        __bf16*       hout = (t & 1) ? hb0 : hb1;
        lstm_step_kernel<<<dim3(HH / 16), dim3(128), 0, stream>>>(
            xet, hin, hout, hf, cf, Wp, bg, bi, bfr, bo);
    }

    // 5) projection + log_softmax
    head_kernel<<<dim3(1), dim3(128), 0, stream>>>(hf, Wph, bp, out);
}
